// TEPC_1D_53111565582337
// MI455X (gfx1250) — compile-verified, matrix operand order checked by reading, not run on hardware
//
#include <hip/hip_runtime.h>
#include <hip/hip_bf16.h>

// CDNA5 / gfx1250 wave32 WMMA types
typedef __attribute__((ext_vector_type(16))) __bf16 v16bf;
typedef __attribute__((ext_vector_type(8)))  float  v8f;
typedef __attribute__((ext_vector_type(4)))  unsigned int u32x4;
typedef __attribute__((ext_vector_type(8)))  unsigned int u32x8;

#define BATCH   128
#define CH      64
#define NDIM    4096
#define NMODES  2049          // NDIM/2 + 1
#define KSTACK  128           // [Re(64) | Im(64)]
#define FFT_THREADS 512

// ---------------------------------------------------------------------------
// Tensor Data Mover: 1-D tile copy global -> LDS (nwords 4-byte elements).
// D# packed per CDNA5 ISA ch.8 (group0 128b, group1 256b, groups2/3 zero).
// Issued once per wave; tracked with TENSORcnt.
// ---------------------------------------------------------------------------
__device__ __forceinline__ void tdm_load_1d(unsigned lds_off, const void* gsrc,
                                            unsigned nwords) {
  const unsigned long long ga = (unsigned long long)(uintptr_t)gsrc;
  u32x4 g0;
  g0[0] = 1u;                                             // count=1, user D#
  g0[1] = lds_off;                                        // lds_addr (bytes)
  g0[2] = (unsigned)(ga & 0xffffffffu);                   // global_addr[31:0]
  g0[3] = (unsigned)((ga >> 32) & 0x1ffffffu) | (2u << 30); // addr[56:32]|type=2
  u32x8 g1;
  g1[0] = (2u << 16);                 // workgroup_mask=0, data_size=2 (4 bytes)
  g1[1] = (nwords & 0xffffu) << 16;   // tensor_dim0[15:0]
  g1[2] = ((nwords >> 16) & 0xffffu)  // tensor_dim0[31:16]
        | (1u << 16);                 // tensor_dim1 = 1
  g1[3] = (nwords & 0xffffu) << 16;   // tensor_dim1[31:16]=0, tile_dim0 = nwords
  g1[4] = 1u;                         // tile_dim1 = 1, tile_dim2 = 0
  g1[5] = nwords;                     // tensor_dim0_stride[31:0]
  g1[6] = 0u;                         // stride hi / tensor_dim1_stride lo
  g1[7] = 0u;                         // tensor_dim1_stride hi
  const u32x4 gz = {0u, 0u, 0u, 0u};  // groups 2/3: dims unused (zero)
  asm volatile("tensor_load_to_lds %0, %1, %2, %3"
               :: "s"(g0), "s"(g1), "s"(gz), "s"(gz)
               : "memory");
}

// ---------------------------------------------------------------------------
// 4096-point radix-2 DIT FFT in LDS. Input must be pre-permuted (bit-reversed
// placement); output natural order. DIR = -1 forward, +1 inverse (unscaled).
// ---------------------------------------------------------------------------
__device__ __forceinline__ unsigned rev12(unsigned v) {
  return __builtin_bitreverse32(v) >> 20;   // 12-bit reversal
}

template <int DIR>
__device__ __forceinline__ void fft4096(float2* s) {
  const int tid = threadIdx.x;
#pragma unroll 1
  for (int st = 1; st <= 12; ++st) {
    __syncthreads();
    const int half = 1 << (st - 1);
    const int m    = half << 1;
    for (int t = tid; t < 2048; t += FFT_THREADS) {
      const int j  = t & (half - 1);
      const int i0 = ((t >> (st - 1)) << st) | j;
      const int i1 = i0 + half;
      const float ang = (float)DIR * 6.28318530717958647692f * (float)j / (float)m;
      float sn, cs;
      __sincosf(ang, &sn, &cs);
      const float2 a = s[i0];
      const float2 b = s[i1];
      const float2 wb = { b.x * cs - b.y * sn, b.x * sn + b.y * cs };
      s[i0] = { a.x + wb.x, a.y + wb.y };
      s[i1] = { a.x - wb.x, a.y - wb.y };
    }
  }
  __syncthreads();
}

// ---------------------------------------------------------------------------
// Kernel 1: realify + transpose weights per mode.
//   Bt[mode][n][k] (bf16, K-contiguous), n<64 -> Re-output column of o=n:
//     k<64: Re w[k][o]   k>=64: -Im w[k-64][o]
//   n>=64 -> Im-output column of o=n-64:
//     k<64: Im w[k][o]   k>=64:  Re w[k-64][o]
// grid (64 o, 33 mode-tiles) x 256
// ---------------------------------------------------------------------------
__global__ __launch_bounds__(256) void wprep_kernel(const float* __restrict__ wr_g,
                                                    const float* __restrict__ wi_g,
                                                    __bf16* __restrict__ Bt) {
  __shared__ float wr[64][65];
  __shared__ float wi[64][65];
  const int o  = blockIdx.x;
  const int x0 = blockIdx.y * 64;
  const int t  = threadIdx.x;

  const int xl_ld = t & 63;
  const int x_ld  = x0 + xl_ld;
  for (int i = t >> 6; i < 64; i += 4) {
    float vr = 0.f, vi = 0.f;
    if (x_ld < NMODES) {
      const size_t base = ((size_t)i * CH + o) * NMODES + x_ld;
      vr = wr_g[base];
      vi = wi_g[base];
    }
    wr[i][xl_ld] = vr;
    wi[i][xl_ld] = vi;
  }
  __syncthreads();

  const int k = t & 127;
  for (int xl = t >> 7; xl < 64; xl += 2) {
    const int x = x0 + xl;
    if (x >= NMODES) break;
    const float v1 = (k < 64) ? wr[k][xl] : -wi[k - 64][xl];
    const float v2 = (k < 64) ? wi[k][xl] :  wr[k - 64][xl];
    __bf16* dst = Bt + (size_t)x * (KSTACK * KSTACK);
    dst[(size_t)o * KSTACK + k]        = (__bf16)v1;
    dst[(size_t)(o + 64) * KSTACK + k] = (__bf16)v2;
  }
}

// ---------------------------------------------------------------------------
// Kernel 2: forward rFFT per (b, in_ch) row + phase derotation -> bf16 A'.
//   zA[mode][b][k] with k<64 = Re z[b][k], k>=64 = Im z[b][k-64]
// grid 8192 x 512
// ---------------------------------------------------------------------------
__global__ __launch_bounds__(FFT_THREADS) void fwd_kernel(const float* __restrict__ x,
                                                          __bf16* __restrict__ zA,
                                                          float* __restrict__ phiW) {
  __shared__ float2 sh[NDIM];
  __shared__ float  phish;
  const int row = blockIdx.x;       // b*64 + i
  const int b   = row >> 6;
  const int ich = row & 63;

  const float* xr = x + (size_t)row * NDIM;
  for (int j = threadIdx.x; j < NDIM; j += FFT_THREADS) {
    sh[rev12(j)] = make_float2(xr[j], 0.f);
  }
  fft4096<-1>(sh);                  // leading __syncthreads covers the load

  if (threadIdx.x == 0) {
    const float ph = atan2f(sh[1].y, sh[1].x);
    phish = ph;
    phiW[row] = ph;
  }
  __syncthreads();
  const float phi = phish;

  for (int m2 = threadIdx.x; m2 < NMODES; m2 += FFT_THREADS) {
    const int k = (m2 <= 1024) ? m2 : m2 - NMODES;   // fftfreq(2049)*2049
    float sn, cs;
    sincosf(-phi * (float)k, &sn, &cs);
    const float zr = sh[m2].x * cs - sh[m2].y * sn;
    const float zi = sh[m2].x * sn + sh[m2].y * cs;
    __bf16* dst = zA + ((size_t)m2 * BATCH + b) * KSTACK;
    dst[ich]      = (__bf16)zr;
    dst[64 + ich] = (__bf16)zi;
  }
}

// ---------------------------------------------------------------------------
// Kernel 3: per-mode 128x128x128 bf16 GEMM via v_wmma_f32_16x16x32_bf16.
// TDM stages the contiguous 32 KB A' and B tiles into LDS (one DMA each,
// issued by wave 0, TENSORcnt-tracked), then 8 waves consume fragments via
// ds_load_b128. Epilogue adds bias, scatters to (b, o, mode) f32 planes.
// grid 2049 x 256
// ---------------------------------------------------------------------------
__global__ __launch_bounds__(256) void gemm_kernel(const __bf16* __restrict__ zA,
                                                   const __bf16* __restrict__ Bt,
                                                   const float* __restrict__ brl,
                                                   const float* __restrict__ bim,
                                                   float* __restrict__ outR,
                                                   float* __restrict__ outI) {
  __shared__ __align__(16) __bf16 Asm[BATCH * KSTACK];   // 32 KB
  __shared__ __align__(16) __bf16 Bsm[KSTACK * KSTACK];  // 32 KB

  const int mode = blockIdx.x;
  const int wave = threadIdx.x >> 5;
  const int lane = threadIdx.x & 31;
  const int lhi  = lane >> 4;         // 0: lanes 0-15, 1: lanes 16-31
  const int l15  = lane & 15;

  if (threadIdx.x < 32) {             // wave 0 issues both tile DMAs
    tdm_load_1d((unsigned)(uintptr_t)&Asm[0],
                zA + (size_t)mode * (BATCH * KSTACK), (BATCH * KSTACK) / 2);
    tdm_load_1d((unsigned)(uintptr_t)&Bsm[0],
                Bt + (size_t)mode * (KSTACK * KSTACK), (KSTACK * KSTACK) / 2);
    __builtin_amdgcn_s_wait_tensorcnt(0);
  }
  __syncthreads();                    // publish LDS tiles to all 8 waves

  union Frag { v16bf v; uint4 u[2]; };

  // Hoist the 4 A fragments (K steps 0,32,64,96) for this wave's M-tile.
  // ISA A-layout (16-bit, 16x32): lanes<16 hold K 0-7 & 16-23, lanes>=16
  // hold K 8-15 & 24-31 for row M = lane&15.
  Frag a[4];
  const int mrow = wave * 16 + l15;
#pragma unroll
  for (int ks = 0; ks < 4; ++ks) {
    const int kb = ks * 32 + lhi * 8;
    a[ks].u[0] = *(const uint4*)(Asm + mrow * KSTACK + kb);
    a[ks].u[1] = *(const uint4*)(Asm + mrow * KSTACK + kb + 16);
  }

#pragma unroll
  for (int nt = 0; nt < 8; ++nt) {
    const int nrow = nt * 16 + l15;   // Bt stored transposed: same pattern as A
    v8f c = {0.f, 0.f, 0.f, 0.f, 0.f, 0.f, 0.f, 0.f};
#pragma unroll
    for (int ks = 0; ks < 4; ++ks) {
      Frag bf;
      const int kb = ks * 32 + lhi * 8;
      bf.u[0] = *(const uint4*)(Bsm + nrow * KSTACK + kb);
      bf.u[1] = *(const uint4*)(Bsm + nrow * KSTACK + kb + 16);
      c = __builtin_amdgcn_wmma_f32_16x16x32_bf16(false, a[ks].v, false, bf.v,
                                                  (short)0, c, false, false);
    }
    // C layout: lane->N (=nrow), VGPR j -> M = 16*wave + j (+8 for lanes>=16)
    const int n    = nrow;
    const float bias = (n < 64) ? brl[(size_t)n * NMODES + mode]
                                : bim[(size_t)(n - 64) * NMODES + mode];
    float* outp  = (n < 64) ? outR : outI;
    const int o  = (n < 64) ? n : n - 64;
#pragma unroll
    for (int j = 0; j < 8; ++j) {
      const int m = wave * 16 + lhi * 8 + j;
      outp[((size_t)m * CH + o) * NMODES + mode] = c[j] + bias;
    }
  }
}

// ---------------------------------------------------------------------------
// Kernel 4: re-rotation + Hermitian irFFT(4096) per (b, out_ch) row.
// grid 8192 x 512
// ---------------------------------------------------------------------------
__global__ __launch_bounds__(FFT_THREADS) void inv_kernel(const float* __restrict__ outR,
                                                          const float* __restrict__ outI,
                                                          const float* __restrict__ phiW,
                                                          float* __restrict__ y) {
  __shared__ float2 sh[NDIM];
  const int row = blockIdx.x;        // b*64 + o
  const int b   = row >> 6;
  const float phi0 = phiW[(size_t)b * CH];   // phase of input channel 0

  const float* rr = outR + (size_t)row * NMODES;
  const float* ri = outI + (size_t)row * NMODES;
  for (int m2 = threadIdx.x; m2 < NMODES; m2 += FFT_THREADS) {
    const int k = (m2 <= 1024) ? m2 : m2 - NMODES;
    float sn, cs;
    sincosf(phi0 * (float)k, &sn, &cs);
    const float re = rr[m2], im = ri[m2];
    const float zr = re * cs - im * sn;
    const float zi = re * sn + im * cs;
    sh[rev12((unsigned)m2)] = make_float2(zr, zi);
    if (m2 >= 1 && m2 <= 2047) {                       // Hermitian mirror
      sh[rev12((unsigned)(NDIM - m2))] = make_float2(zr, -zi);
    }
  }
  fft4096<1>(sh);

  const float scale = 1.0f / (float)NDIM;
  float* yo = y + (size_t)row * NDIM;
  for (int j = threadIdx.x; j < NDIM; j += FFT_THREADS) {
    yo[j] = sh[j].x * scale;   // Re part: DC/Nyquist imag auto-ignored
  }
}

// ---------------------------------------------------------------------------
// Host launcher. Workspace layout (needs ~269 MB):
//   [0]            zA   bf16 2049*128*128
//   [+67,141,632]  Bt   bf16 2049*128*128
//   [+134,283,264] outR f32  128*64*2049
//   [+201,424,896] outI f32  128*64*2049
//   [+268,566,528] phi  f32  8192
// ---------------------------------------------------------------------------
extern "C" void kernel_launch(void* const* d_in, const int* in_sizes, int n_in,
                              void* d_out, int out_size, void* d_ws, size_t ws_size,
                              hipStream_t stream) {
  const float* x      = (const float*)d_in[0];
  const float* w_real = (const float*)d_in[1];
  const float* w_imag = (const float*)d_in[2];
  const float* b_real = (const float*)d_in[3];
  const float* b_imag = (const float*)d_in[4];
  float* out = (float*)d_out;

  char* ws = (char*)d_ws;
  const size_t SZ = (size_t)NMODES * 128 * 128 * 2;   // 67,141,632 bytes
  __bf16* zA  = (__bf16*)(ws);
  __bf16* Bt  = (__bf16*)(ws + SZ);
  float*  oR  = (float*)(ws + 2 * SZ);
  float*  oI  = (float*)(ws + 3 * SZ);
  float*  phi = (float*)(ws + 4 * SZ);

  wprep_kernel<<<dim3(64, 33), 256, 0, stream>>>(w_real, w_imag, Bt);
  fwd_kernel  <<<BATCH * CH, FFT_THREADS, 0, stream>>>(x, zA, phi);
  gemm_kernel <<<NMODES, 256, 0, stream>>>(zA, Bt, b_real, b_imag, oR, oI);
  inv_kernel  <<<BATCH * CH, FFT_THREADS, 0, stream>>>(oR, oI, phi, out);
}
